// rpn_head_30940944400981
// MI455X (gfx1250) — matrix operand order repaired and checked by reading, not compiled, not run on hardware
//
#include <hip/hip_runtime.h>
#include <hip/hip_bf16.h>

typedef __attribute__((ext_vector_type(16))) _Float16 v16h;
typedef __attribute__((ext_vector_type(8)))  float    v8f;
typedef unsigned long long u64;
typedef unsigned int u32;

#define K9      9216
#define NSTAGES 288      /* K9/32 */
#define PRE     6000
#define POST    1000
#define NW      188      /* ceil(6000/32) */
#define NSORT   65536
#define NSCORES 61440    /* 64*64*15 */

// 15 anchors (ratio-major, 5 scales each), derived from the reference generator.
__constant__ float c_anchors[60] = {
  -15.f,  -4.f,  30.f,  19.f,   -38.f, -16.f,  53.f,  31.f,
  -84.f, -40.f,  99.f,  55.f,  -176.f, -88.f, 191.f, 103.f,
 -360.f,-184.f, 375.f, 199.f,
   -8.f,  -8.f,  23.f,  23.f,   -24.f, -24.f,  39.f,  39.f,
  -56.f, -56.f,  71.f,  71.f,  -120.f,-120.f, 135.f, 135.f,
 -248.f,-248.f, 263.f, 263.f,
   -3.f, -14.f,  18.f,  29.f,   -14.f, -36.f,  29.f,  51.f,
  -36.f, -80.f,  51.f,  95.f,   -80.f,-168.f,  95.f, 183.f,
 -168.f,-344.f, 183.f, 359.f
};

// ---- CDNA5 async global->LDS copy (ASYNCcnt-tracked DMA path) ----
__device__ __forceinline__ void async_b128(u32 lds_off, const void* gaddr) {
  asm volatile("global_load_async_to_lds_b128 %0, %1, off"
               :: "v"(lds_off), "v"((unsigned long long)(uintptr_t)gaddr)
               : "memory");
}
__device__ __forceinline__ void wait_async5() {
  asm volatile("s_wait_asynccnt 0x5" ::: "memory");
}
__device__ __forceinline__ void wait_async0() {
  asm volatile("s_wait_asynccnt 0x0" ::: "memory");
}

// ---------------- input packing ----------------
__global__ void k_pack_x(const float* __restrict__ x, _Float16* __restrict__ xh) {
  int t = blockIdx.x * 256 + threadIdx.x;           // 66*66*1024 total
  int c = t & 1023;
  int r = t >> 10;
  int xx = r % 66, yy = r / 66;
  float v = 0.f;
  if (xx >= 1 && xx <= 64 && yy >= 1 && yy <= 64)
    v = x[(size_t)c * 4096 + (size_t)(yy - 1) * 64 + (xx - 1)];
  xh[t] = (_Float16)v;
}

// w_conv (O,I,3,3) f32 -> Bh[o][kq*1024+c] f16  (K-contiguous per output channel)
__global__ void k_pack_w(const float* __restrict__ w, _Float16* __restrict__ Bh) {
  int t = blockIdx.x * 256 + threadIdx.x;           // 1024*9216 total
  int kf = t % K9;
  int o  = t / K9;
  int kq = kf >> 10;
  int c  = kf & 1023;
  Bh[t] = (_Float16)w[((size_t)o * 1024 + c) * 9 + kq];
}

// ---------------- 3x3 conv as implicit GEMM via WMMA ----------------
// Workgroup: 256 thr = 8 waves, tile M=64 (one image row) x N=256.
// Wave tile 32x64: 2 A-frags x 4 B-frags -> 8 WMMA per K-step (1.5 ds_b128/WMMA).
// Double-buffered LDS filled by async global->LDS DMA, 2-stage prefetch depth.
__launch_bounds__(256)
__global__ void k_conv3(const _Float16* __restrict__ xh, const _Float16* __restrict__ Bh,
                        const float* __restrict__ bconv, _Float16* __restrict__ feat) {
  __shared__ _Float16 Ash[2][64 * 40];    // 64 spatial rows x 32K (+8 pad) per buffer
  __shared__ _Float16 Bsh[2][256 * 40];   // 256 out-chan   x 32K (+8 pad) per buffer
  const int t    = threadIdx.x;
  const int lane = t & 31;
  const int wv   = t >> 5;      // 8 waves
  const int wm   = wv & 1;      // wave M sub-tile (32 rows)
  const int wn   = wv >> 1;     // wave N sub-tile (64 cols), 0..3
  const int y    = blockIdx.x;  // one image row per workgroup (M tile = x 0..63)
  const int n0   = blockIdx.y * 256;

  v8f acc[8] = {};              // [0..3]: rows wm*32..+15, [4..7]: rows wm*32+16..+31

  const int arow = t >> 2, aseg = t & 3;   // A stage: 64 rows x 4x16B (1 b128/thr)
                                           // B stage: 256 rows x 64B  (4 b128/thr)
  const u32 aoff[2] = { (u32)(uintptr_t)&Ash[0][arow * 40 + aseg * 8],
                        (u32)(uintptr_t)&Ash[1][arow * 40 + aseg * 8] };
  const u32 boff[2] = { (u32)(uintptr_t)&Bsh[0][t * 40],
                        (u32)(uintptr_t)&Bsh[1][t * 40] };

  auto issue = [&](int stage, int buf) {
    const int kk = stage * 32;
    const int kq = kk >> 10;                 // 0..8 -> (dy,dx)
    const int dy = kq / 3, dx = kq - dy * 3;
    const int c0 = kk & 1023;
    const _Float16* asrc = xh + ((size_t)(y + dy) * 66 + (arow + dx)) * 1024 + c0 + aseg * 8;
    async_b128(aoff[buf], asrc);
    const _Float16* bsrc = Bh + (size_t)(n0 + t) * K9 + kk;   // full 64B row per thread
    async_b128(boff[buf] +  0u, bsrc);
    async_b128(boff[buf] + 16u, bsrc + 8);
    async_b128(boff[buf] + 32u, bsrc + 16);
    async_b128(boff[buf] + 48u, bsrc + 24);
  };

  issue(0, 0);
  issue(1, 1);

  const int hi = lane >> 4, l15 = lane & 15;
  const int ar0 = wm * 32 + l15;             // A frag 0 rows
  const int ar1 = wm * 32 + 16 + l15;        // A frag 1 rows
  const int kb  = hi * 16;                   // B frag K half

  for (int i = 0; i < NSTAGES; ++i) {
    const int buf = i & 1;
    if (i + 1 < NSTAGES) wait_async5(); else wait_async0();  // stage i landed in LDS
    __syncthreads();                                         // ... for every wave

    union Fr { uint4 q[2]; v16h v; };
    const _Float16* A = Ash[buf];
    const _Float16* B = Bsh[buf];
    Fr fa0, fa1;
    // A: lanes 0-15 hold K{0..7,16..23}, lanes 16-31 K{8..15,24..31}
    fa0.q[0] = *(const uint4*)&A[ar0 * 40 + hi * 8];
    fa0.q[1] = *(const uint4*)&A[ar0 * 40 + 16 + hi * 8];
    fa1.q[0] = *(const uint4*)&A[ar1 * 40 + hi * 8];
    fa1.q[1] = *(const uint4*)&A[ar1 * 40 + 16 + hi * 8];
#pragma unroll
    for (int nb = 0; nb < 4; ++nb) {
      Fr fb;                                 // B: col = lane&15, same K order as A
      const int col = wn * 64 + nb * 16 + l15;
      fb.q[0] = *(const uint4*)&B[col * 40 + hi * 8];
      fb.q[1] = *(const uint4*)&B[col * 40 + 16 + hi * 8];
      acc[nb]     = __builtin_amdgcn_wmma_f32_16x16x32_f16(false, fa0.v, false, fb.v, (short)0, acc[nb],     false, false);
      acc[4 + nb] = __builtin_amdgcn_wmma_f32_16x16x32_f16(false, fa1.v, false, fb.v, (short)0, acc[4 + nb], false, false);
    }

    __syncthreads();                        // all waves done reading this buffer
    if (i + 2 < NSTAGES) issue(i + 2, buf); // refill it two stages ahead
  }

  // D layout: M = (lane>=16)*8 + vgpr, N = lane&15. Fuse bias + ReLU, store HWC f16.
#pragma unroll
  for (int g = 0; g < 2; ++g) {
    const int xbase = wm * 32 + g * 16 + hi * 8;
#pragma unroll
    for (int nb = 0; nb < 4; ++nb) {
      const int n  = n0 + wn * 64 + nb * 16 + l15;
      const float bc = bconv[n];
#pragma unroll
      for (int v = 0; v < 8; ++v) {
        float val = acc[g * 4 + nb][v] + bc;
        val = val > 0.f ? val : 0.f;
        feat[((size_t)y * 64 + (xbase + v)) * 1024 + n] = (_Float16)val;
      }
    }
  }
}

// ---------------- 1x1 heads: 15 sigmoid scores (as sort keys) + 60 deltas ----------------
__launch_bounds__(128)
__global__ void k_heads(const _Float16* __restrict__ feat,
                        const float* __restrict__ wcls, const float* __restrict__ bcls,
                        const float* __restrict__ wbbox, const float* __restrict__ bbbox,
                        u64* __restrict__ keys, float* __restrict__ deltas) {
  __shared__ _Float16 sf[1024];
  const int p = blockIdx.x;      // spatial position 0..4095
  const int t = threadIdx.x;
  {
    uint4 v = *(const uint4*)(feat + (size_t)p * 1024 + t * 8);
    *(uint4*)(&sf[t * 8]) = v;
  }
  __syncthreads();
  if (t < 75) {
    const float* wrow = (t < 15) ? (wcls + (size_t)t * 1024) : (wbbox + (size_t)(t - 15) * 1024);
    float sum = 0.f;
    for (int c = 0; c < 1024; ++c) sum += (float)sf[c] * wrow[c];
    if (t < 15) {
      sum += bcls[t];
      float s = 1.f / (1.f + expf(-sum));
      u32 q  = (u32)p * 15 + (u32)t;
      u32 ub = __float_as_uint(s);
      u32 kb = (ub & 0x80000000u) ? ~ub : (ub | 0x80000000u);  // order-preserving
      keys[q] = ((u64)kb << 32) | (u64)(0xFFFFFFFFu - q);      // desc score, asc index
    } else {
      int oo = t - 15;
      deltas[(size_t)p * 60 + oo] = sum + bbbox[oo];
    }
  }
}

__global__ void k_padkeys(u64* __restrict__ keys) {
  int i = blockIdx.x * 256 + threadIdx.x;      // 4096 pads
  u32 q = (u32)(NSCORES + i);
  u32 ub = __float_as_uint(-1.0f);
  u32 kb = ~ub;                                 // negative -> flip all bits
  keys[q] = ((u64)kb << 32) | (u64)(0xFFFFFFFFu - q);
}

// ---------------- bitonic sort stage (descending) ----------------
__global__ void k_bitonic(u64* __restrict__ keys, u32 j, u32 k) {
  u32 i = blockIdx.x * 256 + threadIdx.x;
  u32 ixj = i ^ j;
  if (ixj > i) {
    u64 a = keys[i], b = keys[ixj];
    bool up = ((i & k) == 0);                   // "up" region sorts descending
    if (up ? (a < b) : (a > b)) { keys[i] = b; keys[ixj] = a; }
  }
}

// ---------------- box decode + clip + min-size ----------------
__global__ void k_decode(const u64* __restrict__ keys, const float* __restrict__ deltas,
                         const int* __restrict__ hp, const int* __restrict__ wp,
                         const int* __restrict__ sfp,
                         float* __restrict__ boxes, float* __restrict__ tsc) {
  int i = blockIdx.x * 256 + threadIdx.x;
  if (i >= PRE) return;
  u64 key = keys[i];
  u32 q  = 0xFFFFFFFFu - (u32)(key & 0xFFFFFFFFull);
  u32 sb = (u32)(key >> 32);
  float score = __uint_as_float((sb & 0x80000000u) ? (sb & 0x7FFFFFFFu) : ~sb);
  int a = (int)(q % 15u), p = (int)(q / 15u);
  float shx = (float)((p & 63) * 16);
  float shy = (float)((p >> 6) * 16);
  float ax1 = c_anchors[a * 4 + 0] + shx, ay1 = c_anchors[a * 4 + 1] + shy;
  float ax2 = c_anchors[a * 4 + 2] + shx, ay2 = c_anchors[a * 4 + 3] + shy;
  float wdt = ax2 - ax1 + 1.f, hgt = ay2 - ay1 + 1.f;
  float cx = ax1 + 0.5f * wdt, cy = ay1 + 0.5f * hgt;
  const float* d = deltas + (size_t)p * 60 + a * 4;
  const float BXC = 4.135166556742356f;  // log(1000/16)
  float dx = d[0], dyv = d[1];
  float dw = fminf(d[2], BXC), dh = fminf(d[3], BXC);
  float pcx = dx * wdt + cx, pcy = dyv * hgt + cy;
  float pw = expf(dw) * wdt, ph = expf(dh) * hgt;
  float wf = (float)wp[0] - 1.f, hf = (float)hp[0] - 1.f;
  float x1 = fminf(fmaxf(pcx - 0.5f * pw, 0.f), wf);
  float y1 = fminf(fmaxf(pcy - 0.5f * ph, 0.f), hf);
  float x2 = fminf(fmaxf(pcx + 0.5f * pw - 1.f, 0.f), wf);
  float y2 = fminf(fmaxf(pcy + 0.5f * ph - 1.f, 0.f), hf);
  boxes[i * 4 + 0] = x1; boxes[i * 4 + 1] = y1;
  boxes[i * 4 + 2] = x2; boxes[i * 4 + 3] = y2;
  float ms = 0.0f * (float)sfp[0];  // MIN_SIZE = 0
  bool valid = (x2 - x1 + 1.f >= ms) && (y2 - y1 + 1.f >= ms);
  tsc[i] = valid ? score : -__builtin_inff();
}

// ---------------- NMS suppression bitmask (bits only for j > i) ----------------
__global__ void k_mask(const float* __restrict__ boxes, u32* __restrict__ mask) {
  int tid = blockIdx.x * 256 + threadIdx.x;
  if (tid >= PRE * NW) return;
  int i = tid / NW;
  int w = tid - i * NW;
  float x1 = boxes[i * 4 + 0], y1 = boxes[i * 4 + 1];
  float x2 = boxes[i * 4 + 2], y2 = boxes[i * 4 + 3];
  float ai = (x2 - x1 + 1.f) * (y2 - y1 + 1.f);
  u32 bits = 0;
  int jbase = w * 32;
  for (int b = 0; b < 32; ++b) {
    int j = jbase + b;
    if (j >= PRE || j <= i) continue;
    float bx1 = boxes[j * 4 + 0], by1 = boxes[j * 4 + 1];
    float bx2 = boxes[j * 4 + 2], by2 = boxes[j * 4 + 3];
    float aj = (bx2 - bx1 + 1.f) * (by2 - by1 + 1.f);
    float xx1 = fmaxf(x1, bx1), yy1 = fmaxf(y1, by1);
    float xx2 = fminf(x2, bx2), yy2 = fminf(y2, by2);
    float iw = fmaxf(xx2 - xx1 + 1.f, 0.f), ih = fmaxf(yy2 - yy1 + 1.f, 0.f);
    float inter = iw * ih;
    float iou = inter / (ai + aj - inter);
    if (iou > 0.7f) bits |= (1u << b);
  }
  mask[tid] = bits;
}

// ---------------- greedy scan (single wave, lockstep) + compact into ROIs ----------------
__launch_bounds__(256)
__global__ void k_final(const u32* __restrict__ mask, const float* __restrict__ boxes,
                        const float* __restrict__ tsc, float* __restrict__ out) {
  __shared__ u32 keep[NW];
  __shared__ u32 kf[NW];
  __shared__ int pref[NW];
  const int t = threadIdx.x;
  for (int i = t; i < POST * 5; i += 256) out[i] = 0.f;
  if (t < NW) keep[t] = 0xFFFFFFFFu;
  __syncthreads();
  if (t < 32) {
    volatile u32* vk = keep;
    for (int i = 0; i < PRE; ++i) {
      u32 kword = vk[i >> 5];
      if (kword & (1u << (i & 31))) {
        const u32* mrow = mask + (size_t)i * NW;
        for (int w = t; w < NW; w += 32) {
          u32 nv = vk[w] & ~mrow[w];
          vk[w] = nv;
        }
      }
    }
  }
  __syncthreads();
  if (t < NW) {
    u32 bits = keep[t];
    for (int b = 0; b < 32; ++b) {
      int j = t * 32 + b;
      if (j >= PRE || !(tsc[j] > -1e37f)) bits &= ~(1u << b);
    }
    kf[t] = bits;
  }
  __syncthreads();
  if (t == 0) {
    int r = 0;
    for (int w = 0; w < NW; ++w) { pref[w] = r; r += __popc(kf[w]); }
  }
  __syncthreads();
  if (t < NW) {
    int r = pref[t];
    u32 bits = kf[t];
    while (bits) {
      int b = __ffs(bits) - 1;
      bits &= bits - 1;
      int j = t * 32 + b;
      if (r < POST) {
        out[r * 5 + 0] = 0.f;
        out[r * 5 + 1] = boxes[j * 4 + 0];
        out[r * 5 + 2] = boxes[j * 4 + 1];
        out[r * 5 + 3] = boxes[j * 4 + 2];
        out[r * 5 + 4] = boxes[j * 4 + 3];
      }
      ++r;
    }
  }
}

extern "C" void kernel_launch(void* const* d_in, const int* in_sizes, int n_in,
                              void* d_out, int out_size, void* d_ws, size_t ws_size,
                              hipStream_t stream) {
  (void)in_sizes; (void)n_in; (void)out_size; (void)ws_size;
  const float* x_in  = (const float*)d_in[0];
  const float* wconv = (const float*)d_in[1];
  const float* bconv = (const float*)d_in[2];
  const float* wcls  = (const float*)d_in[3];
  const float* bcls  = (const float*)d_in[4];
  const float* wbbox = (const float*)d_in[5];
  const float* bbbox = (const float*)d_in[6];
  const int*   hp    = (const int*)d_in[7];
  const int*   wp    = (const int*)d_in[8];
  const int*   sfp   = (const int*)d_in[9];
  float* out = (float*)d_out;

  char* ws = (char*)d_ws;
  size_t off = 0;
  auto alloc = [&](size_t bytes) -> void* {
    off = (off + 255) & ~(size_t)255;
    void* p = ws + off;
    off += bytes;
    return p;
  };
  _Float16* xh   = (_Float16*)alloc((size_t)66 * 66 * 1024 * 2);   // padded HWC input
  _Float16* Bh   = (_Float16*)alloc((size_t)1024 * K9 * 2);        // packed conv weights
  _Float16* feat = (_Float16*)alloc((size_t)4096 * 1024 * 2);      // conv output HWC
  u64*      keys = (u64*)alloc((size_t)NSORT * 8);                 // sort keys
  float*  deltas = (float*)alloc((size_t)4096 * 60 * 4);
  float*   boxes = (float*)alloc((size_t)PRE * 4 * 4);
  float*     tsc = (float*)alloc((size_t)PRE * 4);
  u32*      mask = (u32*)alloc((size_t)PRE * NW * 4);

  k_pack_x<<<(66 * 66 * 1024) / 256, 256, 0, stream>>>(x_in, xh);
  k_pack_w<<<(1024 * K9) / 256, 256, 0, stream>>>(wconv, Bh);
  k_conv3<<<dim3(64, 4), 256, 0, stream>>>(xh, Bh, bconv, feat);
  k_heads<<<4096, 128, 0, stream>>>(feat, wcls, bcls, wbbox, bbbox, keys, deltas);
  k_padkeys<<<(NSORT - NSCORES) / 256, 256, 0, stream>>>(keys);
  for (u32 k = 2; k <= (u32)NSORT; k <<= 1)
    for (u32 j = k >> 1; j > 0; j >>= 1)
      k_bitonic<<<NSORT / 256, 256, 0, stream>>>(keys, j, k);
  k_decode<<<(PRE + 255) / 256, 256, 0, stream>>>(keys, deltas, hp, wp, sfp, boxes, tsc);
  k_mask<<<(PRE * NW + 255) / 256, 256, 0, stream>>>(boxes, mask);
  k_final<<<1, 256, 0, stream>>>(mask, boxes, tsc, out);
}
